// TemporalGNN_64879775973720
// MI455X (gfx1250) — compile-verified, matrix operand order checked by reading, not run on hardware
//
#include <hip/hip_runtime.h>
#include <math.h>

typedef __attribute__((ext_vector_type(16))) _Float16 v16h;
typedef __attribute__((ext_vector_type(8)))  float    v8f;
typedef int v4i __attribute__((vector_size(16)));

#define T_FRAMES 16
#define DIM      256
#define G3       768

// ---------------------------------------------------------------------------
// CDNA5 async global->LDS copy (ASYNCcnt path), with safe fallback.
// Probe round 2 diagnostic: param 0 of the builtin is  v4i __device__ *  (AS1).
// ---------------------------------------------------------------------------
#if defined(__gfx1250__) && __has_builtin(__builtin_amdgcn_global_load_async_to_lds_b128)
#define USE_ASYNC 1
typedef __attribute__((address_space(1))) v4i* gv4i_p;   // global int4*
typedef __attribute__((address_space(3))) v4i* lv4i_p;   // LDS int4*
#else
#define USE_ASYNC 0
#endif

__device__ __forceinline__ void cp16(const _Float16* g, _Float16* l) {
#if USE_ASYNC
  __builtin_amdgcn_global_load_async_to_lds_b128(
      (gv4i_p)(unsigned long long)(const void*)g,
      (lv4i_p)(unsigned)(unsigned long long)(void*)l,
      0, 0);
#else
  *(int4*)l = *(const int4*)g;
#endif
}

__device__ __forceinline__ void cp_wait() {
#if USE_ASYNC
  asm volatile("s_wait_asynccnt 0" ::: "memory");
#endif
}

// ---------------------------------------------------------------------------
// WMMA GEMM, all-f16 operands: C[M x Nc] = A[M x K] * Bt^T (+bias)
//   A : [M][K] f16 row-major,  Bt : [Nc][K] f16 row-major (output-col major).
// Block 256 threads = 8 wave32.  Tile 64(M) x 128(N) x 32(K), LDS double-buffer.
// Wave grid 2x4; each wave owns 32x32 -> 4 accumulators, 4 WMMA / K-step.
// ---------------------------------------------------------------------------
__global__ __launch_bounds__(256) void k_gemm_f16(
    const _Float16* __restrict__ A, const _Float16* __restrict__ Bt,
    float* __restrict__ C, const float* __restrict__ bias,
    int M, int Nc, int K)
{
  // pad rows to 40 halves (80B) -> b128 LDS writes 16B-aligned, K-pair reads 4B-aligned
  __shared__ __align__(16) _Float16 As[2][64][40];
  __shared__ __align__(16) _Float16 Bs[2][128][40];

  const int tid  = threadIdx.x;
  const int lane = tid & 31;
  const int w    = tid >> 5;
  const int wm   = w >> 2;        // 0..1 : 32-row block
  const int wn   = w & 3;         // 0..3 : 32-col block
  const int half = lane >> 4;     // K-half selector (ISA 16-bit frag layout)
  const int l15  = lane & 15;

  const int tile_n = blockIdx.x * 128;
  const int tile_m = blockIdx.y * 64;

  // staging coordinates: one 16B chunk for A, two for B, per thread
  const int am  = tid >> 2;
  const int ak8 = (tid & 3) << 3;
  int agr = tile_m + am; if (agr >= M) agr = M - 1;   // clamp: no divergence
  const _Float16* Arow  = A  + (size_t)agr * K + ak8;

  const int cn0  = tid;                     // chunk ids tid and tid+256
  const int bn_0 = cn0 >> 2,         bk_0 = (cn0 & 3) << 3;
  const int bn_1 = (cn0 + 256) >> 2, bk_1 = ((cn0 + 256) & 3) << 3;
  const _Float16* Brow0 = Bt + (size_t)(tile_n + bn_0) * K + bk_0;
  const _Float16* Brow1 = Bt + (size_t)(tile_n + bn_1) * K + bk_1;

  auto stage = [&](int buf, int kt) {
    cp16(Arow  + kt, &As[buf][am][ak8]);
    cp16(Brow0 + kt, &Bs[buf][bn_0][bk_0]);
    cp16(Brow1 + kt, &Bs[buf][bn_1][bk_1]);
  };

  v8f acc00 = {}, acc01 = {}, acc10 = {}, acc11 = {};

  stage(0, 0);
  cp_wait();
  __syncthreads();

  for (int kt = 0; kt < K; kt += 32) {
    const int cur = (kt >> 5) & 1;
    if (kt + 32 < K) stage(cur ^ 1, kt + 32);

    v16h a0, a1, b0, b1;
    #pragma unroll
    for (int v = 0; v < 8; ++v) {
      // K pair base per ISA 16-bit A/B layout: (v>=4?16:0) + half*8 + 2*(v&3)
      const int k0 = ((v & 4) << 2) + (half << 3) + ((v & 3) << 1);
      union { unsigned u; _Float16 h[2]; } p;
      p.u = *(const unsigned*)&As[cur][wm * 32      + l15][k0]; a0[2*v]=p.h[0]; a0[2*v+1]=p.h[1];
      p.u = *(const unsigned*)&As[cur][wm * 32 + 16 + l15][k0]; a1[2*v]=p.h[0]; a1[2*v+1]=p.h[1];
      p.u = *(const unsigned*)&Bs[cur][wn * 32      + l15][k0]; b0[2*v]=p.h[0]; b0[2*v+1]=p.h[1];
      p.u = *(const unsigned*)&Bs[cur][wn * 32 + 16 + l15][k0]; b1[2*v]=p.h[0]; b1[2*v+1]=p.h[1];
    }
    acc00 = __builtin_amdgcn_wmma_f32_16x16x32_f16(false, a0, false, b0, (short)0, acc00, false, false);
    acc01 = __builtin_amdgcn_wmma_f32_16x16x32_f16(false, a0, false, b1, (short)0, acc01, false, false);
    acc10 = __builtin_amdgcn_wmma_f32_16x16x32_f16(false, a1, false, b0, (short)0, acc10, false, false);
    acc11 = __builtin_amdgcn_wmma_f32_16x16x32_f16(false, a1, false, b1, (short)0, acc11, false, false);

    cp_wait();
    __syncthreads();
  }

  // C/D layout: m = v + 8*half (+16 per M-subtile), n = lane&15 (+16 per N-subtile)
  const int n0 = tile_n + wn * 32 + l15;
  const int n1 = n0 + 16;
  const float bv0 = bias ? bias[n0] : 0.0f;
  const float bv1 = bias ? bias[n1] : 0.0f;
  #pragma unroll
  for (int v = 0; v < 8; ++v) {
    const int m0 = tile_m + wm * 32 + half * 8 + v;
    const int m1 = m0 + 16;
    if (m0 < M) { C[(size_t)m0 * Nc + n0] = acc00[v] + bv0;
                  C[(size_t)m0 * Nc + n1] = acc01[v] + bv1; }
    if (m1 < M) { C[(size_t)m1 * Nc + n0] = acc10[v] + bv0;
                  C[(size_t)m1 * Nc + n1] = acc11[v] + bv1; }
  }
}

// ---------------------------------------------------------------------------
// Conversion / helper kernels
// ---------------------------------------------------------------------------
__global__ void k_cvt_f16(const float* __restrict__ src, _Float16* __restrict__ dst,
                          long long n) {
  long long i = (long long)blockIdx.x * 256 + threadIdx.x;
  if (i < n) dst[i] = (_Float16)src[i];
}

// Wt[c][r] = (f16) W[r][c]   (R x C input)
__global__ void k_transpose_cvt(const float* __restrict__ W, _Float16* __restrict__ Wt,
                                int R, int Ccols) {
  int i = blockIdx.x * 256 + threadIdx.x;
  if (i < R * Ccols) {
    int r = i / Ccols, c = i - r * Ccols;
    Wt[(size_t)c * R + r] = (_Float16)W[i];
  }
}

__global__ void k_fill_f32(float* __restrict__ p, float v, int n) {
  int i = blockIdx.x * 256 + threadIdx.x;
  if (i < n) p[i] = v;
}

__global__ void k_fill_f16(_Float16* __restrict__ p, int n) {
  int i = blockIdx.x * 256 + threadIdx.x;
  if (i < n) p[i] = (_Float16)0.0f;
}

__global__ void k_deg(const int* __restrict__ cols, float* __restrict__ deg, int E) {
  int i = blockIdx.x * 256 + threadIdx.x;
  if (i < E) atomicAdd(&deg[cols[i]], 1.0f);
}

__global__ void k_dinv(const float* __restrict__ deg, float* __restrict__ dinv, int n) {
  int i = blockIdx.x * 256 + threadIdx.x;
  if (i < n) dinv[i] = rsqrtf(deg[i]);      // deg >= 1 (self loop)
}

// self-loop init: acc[i][f] = xw[i][f] * dinv[i]^2
__global__ void k_selfinit(const float* __restrict__ xw, const float* __restrict__ dinv,
                           float* __restrict__ acc) {
  const int node = blockIdx.x, f = threadIdx.x;
  const float dv = dinv[node];
  acc[(size_t)node * DIM + f] = xw[(size_t)node * DIM + f] * dv * dv;
}

// one wave32 per edge; float4 gathers, 8 atomics per lane
__global__ void k_scatter(const float* __restrict__ xw,
                          const int* __restrict__ rows, const int* __restrict__ cols,
                          const float* __restrict__ dinv, float* __restrict__ acc, int E) {
  const int e = blockIdx.x * 8 + (threadIdx.x >> 5);
  const int lane = threadIdx.x & 31;
  if (e >= E) return;
  const int r = rows[e], c = cols[e];
  const float nrm = dinv[r] * dinv[c];
  const float4* src = (const float4*)(xw + (size_t)r * DIM);
  float* dst = acc + (size_t)c * DIM;
  #pragma unroll
  for (int j = 0; j < 2; ++j) {
    const int q = lane + j * 32;          // 64 float4 chunks per row
    float4 v = src[q];
    atomicAdd(dst + q * 4 + 0, v.x * nrm);
    atomicAdd(dst + q * 4 + 1, v.y * nrm);
    atomicAdd(dst + q * 4 + 2, v.z * nrm);
    atomicAdd(dst + q * 4 + 3, v.w * nrm);
  }
}

// bias + ReLU -> f16 frame of H
__global__ void k_relu_to_H(const float* __restrict__ acc, const float* __restrict__ bgcn,
                            _Float16* __restrict__ Ht) {
  const size_t i = (size_t)blockIdx.x * 256 + threadIdx.x;
  float v = acc[i] + bgcn[threadIdx.x & 255];
  Ht[i] = (_Float16)(v > 0.0f ? v : 0.0f);
}

// fused GRU gates (torch order r,z,n): h' = (1-z)*n + z*h   (h kept in f16)
__global__ void k_gru_gate(const float* __restrict__ gi, const float* __restrict__ gh,
                           const _Float16* __restrict__ hprev, _Float16* __restrict__ hnew,
                           int total) {
  const int i = blockIdx.x * 256 + threadIdx.x;
  if (i >= total) return;
  const int n = i >> 8, f = i & 255;
  const float* gin = gi + (size_t)n * G3;
  const float* ghn = gh + (size_t)n * G3;
  const float r  = 1.0f / (1.0f + expf(-(gin[f]       + ghn[f])));
  const float z  = 1.0f / (1.0f + expf(-(gin[256 + f] + ghn[256 + f])));
  const float nn = tanhf(gin[512 + f] + r * ghn[512 + f]);
  hnew[i] = (_Float16)((1.0f - z) * nn + z * (float)hprev[i]);
}

// head: per-node dot(h, W_head) via wave32 shuffle reduction + sigmoid
__global__ void k_head(const _Float16* __restrict__ h, const float* __restrict__ Wh,
                       const float* __restrict__ bh, float* __restrict__ risk, int Nn) {
  const int node = blockIdx.x * 8 + (threadIdx.x >> 5);
  const int lane = threadIdx.x & 31;
  if (node >= Nn) return;
  float s = 0.0f;
  #pragma unroll
  for (int j = 0; j < 8; ++j)
    s += (float)h[(size_t)node * DIM + lane + j * 32] * Wh[lane + j * 32];
  #pragma unroll
  for (int off = 16; off > 0; off >>= 1) s += __shfl_down(s, off, 32);
  if (lane == 0) risk[node] = 1.0f / (1.0f + expf(-(s + bh[0])));
}

__global__ void k_h_to_f32(const _Float16* __restrict__ src, float* __restrict__ dst, int n) {
  int i = blockIdx.x * 256 + threadIdx.x;
  if (i < n) dst[i] = (float)src[i];
}

// ---------------------------------------------------------------------------
extern "C" void kernel_launch(void* const* d_in, const int* in_sizes, int n_in,
                              void* d_out, int out_size, void* d_ws, size_t ws_size,
                              hipStream_t stream) {
  const float* x_seq  = (const float*)d_in[0];   // [T, N, 256]
  const int*   eidx   = (const int*)  d_in[1];   // [T, 2, E]
  const float* W_gcn  = (const float*)d_in[2];   // [256, 256] (K x N)
  const float* b_gcn  = (const float*)d_in[3];
  const float* W_ih   = (const float*)d_in[4];   // [768, 256]
  const float* W_hh   = (const float*)d_in[5];   // [768, 256]
  const float* b_ih   = (const float*)d_in[6];
  const float* b_hh   = (const float*)d_in[7];
  const float* W_head = (const float*)d_in[8];   // [256, 1]
  const float* b_head = (const float*)d_in[9];

  const int N = in_sizes[0] / (T_FRAMES * DIM);   // 20000
  const int E = in_sizes[1] / (T_FRAMES * 2);     // 640000

  // ---- workspace carve-up (256B aligned) ----
  char* base = (char*)d_ws;
  size_t off = 0;
  auto carve = [&](size_t bytes) -> char* {
    char* p = base + off;
    off = (off + bytes + 255) & ~(size_t)255;
    return p;
  };
  _Float16* Xh    = (_Float16*)carve((size_t)T_FRAMES * N * DIM * sizeof(_Float16));
  _Float16* Hbuf  = (_Float16*)carve((size_t)T_FRAMES * N * DIM * sizeof(_Float16));
  _Float16* Wgt   = (_Float16*)carve((size_t)DIM * DIM * sizeof(_Float16));   // [N][K]
  _Float16* Wih16 = (_Float16*)carve((size_t)G3 * DIM * sizeof(_Float16));    // [N][K]
  _Float16* Whh16 = (_Float16*)carve((size_t)G3 * DIM * sizeof(_Float16));    // [N][K]
  float*    xw    = (float*)   carve((size_t)N * DIM * sizeof(float));
  float*    acc   = (float*)   carve((size_t)N * DIM * sizeof(float));
  float*    deg   = (float*)   carve((size_t)N * sizeof(float));
  float*    dinv  = (float*)   carve((size_t)N * sizeof(float));
  float*    gi    = (float*)   carve((size_t)N * G3 * sizeof(float));
  float*    gh    = (float*)   carve((size_t)N * G3 * sizeof(float));
  _Float16* h0    = (_Float16*)carve((size_t)N * DIM * sizeof(_Float16));
  _Float16* h1    = (_Float16*)carve((size_t)N * DIM * sizeof(_Float16));
  (void)ws_size; (void)n_in;

  const dim3 blk(256);
  const long long nX = (long long)T_FRAMES * N * DIM;
  const int nb1  = (N + 255) / 256;
  const int nbE  = (E + 255) / 256;
  const dim3 gGcn(DIM / 128, (N + 63) / 64);     // 2 x 313
  const dim3 gGru(G3  / 128, (N + 63) / 64);     // 6 x 313

  // ---- one-time (per launch) f16 conversions ----
  k_cvt_f16<<<(unsigned)((nX + 255) / 256), blk, 0, stream>>>(x_seq, Xh, nX);
  k_transpose_cvt<<<(DIM * DIM + 255) / 256, blk, 0, stream>>>(W_gcn, Wgt, DIM, DIM);
  k_cvt_f16<<<(G3 * DIM + 255) / 256, blk, 0, stream>>>(W_ih, Wih16, (long long)G3 * DIM);
  k_cvt_f16<<<(G3 * DIM + 255) / 256, blk, 0, stream>>>(W_hh, Whh16, (long long)G3 * DIM);

  // ---- stage 1: per-frame GCN + ReLU -> H (f16) ----
  for (int t = 0; t < T_FRAMES; ++t) {
    const _Float16* Xt   = Xh + (size_t)t * N * DIM;
    const int*      rows = eidx + (size_t)t * 2 * E;
    const int*      cols = rows + E;

    k_gemm_f16<<<gGcn, blk, 0, stream>>>(Xt, Wgt, xw, nullptr, N, DIM, DIM);
    k_fill_f32<<<nb1, blk, 0, stream>>>(deg, 1.0f, N);
    k_deg<<<nbE, blk, 0, stream>>>(cols, deg, E);
    k_dinv<<<nb1, blk, 0, stream>>>(deg, dinv, N);
    k_selfinit<<<N, blk, 0, stream>>>(xw, dinv, acc);
    k_scatter<<<(E + 7) / 8, blk, 0, stream>>>(xw, rows, cols, dinv, acc, E);
    k_relu_to_H<<<N, blk, 0, stream>>>(acc, b_gcn, Hbuf + (size_t)t * N * DIM);
  }

  // ---- stage 2: GRU over T (h0 = 0, hidden kept in f16) ----
  k_fill_f16<<<(N * DIM + 255) / 256, blk, 0, stream>>>(h0, N * DIM);
  _Float16* hp = h0;
  _Float16* hn = h1;
  for (int t = 0; t < T_FRAMES; ++t) {
    const _Float16* Ht = Hbuf + (size_t)t * N * DIM;
    k_gemm_f16<<<gGru, blk, 0, stream>>>(Ht, Wih16, gi, b_ih, N, G3, DIM);
    k_gemm_f16<<<gGru, blk, 0, stream>>>(hp, Whh16, gh, b_hh, N, G3, DIM);
    k_gru_gate<<<(N * DIM + 255) / 256, blk, 0, stream>>>(gi, gh, hp, hn, N * DIM);
    _Float16* tmp = hp; hp = hn; hn = tmp;
  }

  // ---- stage 3: head + outputs ----
  float* out_risk = (float*)d_out;        // [N]
  float* out_h    = (float*)d_out + N;    // [N, 256]
  k_head<<<(N + 7) / 8, blk, 0, stream>>>(hp, W_head, b_head, out_risk, N);
  k_h_to_f32<<<(N * DIM + 255) / 256, blk, 0, stream>>>(hp, out_h, N * DIM);
  (void)out_size;
}
